// LlamaAttention_40913858462370
// MI455X (gfx1250) — matrix operand order repaired and checked by reading, not compiled, not run on hardware
//
#include <hip/hip_runtime.h>
#include <hip/hip_bf16.h>
#include <stdint.h>

// ---------------------------------------------------------------- constants
#define TT   2048
#define DD   4096
#define NH   32
#define KH   8
#define HH   128
#define SLOTS 32768
#define NEG_INF -1e30f

typedef __bf16 v16bf __attribute__((ext_vector_type(16)));
typedef float  v8f  __attribute__((ext_vector_type(8)));

union Frag32 { v16bf v; unsigned int u[8]; };
union U4     { uint4 q; unsigned int u[4]; unsigned short s[8]; };

__device__ __forceinline__ unsigned short f32_to_bf16(float f) {
    unsigned int u = __float_as_uint(f);
    unsigned int r = u + 0x7FFFu + ((u >> 16) & 1u);
    return (unsigned short)(r >> 16);
}

__device__ __forceinline__ v8f wmma_bf16(v16bf a, v16bf b, v8f c) {
    return __builtin_amdgcn_wmma_f32_16x16x32_bf16(false, a, false, b,
                                                   (short)0, c, false, false);
}

// Async DMA: 16B from global to LDS, tracked with ASYNCcnt.
// VDST VGPR = per-lane LDS byte address, VADDR = per-lane 64-bit global addr.
__device__ __forceinline__ void async_load_b128(unsigned lds_off, const unsigned short* gaddr) {
    asm volatile("global_load_async_to_lds_b128 %0, %1, off"
                 :: "v"(lds_off), "v"(gaddr)
                 : "memory");
}
__device__ __forceinline__ void wait_async_le2() {
    asm volatile("s_wait_asynccnt 0x2" ::: "memory");
}
__device__ __forceinline__ void wait_async_le0() {
    asm volatile("s_wait_asynccnt 0x0" ::: "memory");
}

// A-matrix fragment, 16x32 bf16 (M x K). base = tile origin (row 0, k 0),
// stride in ushorts. Lanes 0-15: row=lane, K chunks [0..7] and [16..23];
// lanes 16-31: row=lane-16, K chunks [8..15] and [24..31].
__device__ __forceinline__ v16bf load_frag_a(const unsigned short* base, int stride, int lane) {
    int half = (lane >> 4) & 1;
    int r    = lane & 15;
    const unsigned short* p = base + r * stride + half * 8;
    U4 a, b;
    a.q = *(const uint4*)(p);
    b.q = *(const uint4*)(p + 16);
    Frag32 f;
    f.u[0]=a.u[0]; f.u[1]=a.u[1]; f.u[2]=a.u[2]; f.u[3]=a.u[3];
    f.u[4]=b.u[0]; f.u[5]=b.u[1]; f.u[6]=b.u[2]; f.u[7]=b.u[3];
    return f.v;
}

// B-matrix fragment, 32x16 bf16 (K x N), stored N-major: base[n*stride + k].
// Lanes 0-15: col=lane, K chunks [0..7] and [16..23];
// lanes 16-31: col=lane-16, K chunks [8..15] and [24..31].
__device__ __forceinline__ v16bf load_frag_b(const unsigned short* base, int stride, int lane) {
    int half = (lane >> 4) & 1;
    int r    = lane & 15;
    const unsigned short* p = base + r * stride + half * 8;
    U4 a, b;
    a.q = *(const uint4*)(p);
    b.q = *(const uint4*)(p + 16);
    Frag32 f;
    f.u[0]=a.u[0]; f.u[1]=a.u[1]; f.u[2]=a.u[2]; f.u[3]=a.u[3];
    f.u[4]=b.u[0]; f.u[5]=b.u[1]; f.u[6]=b.u[2]; f.u[7]=b.u[3];
    return f.v;
}

// ---------------------------------------------------------------- cache copy
__global__ void copy_f4_kernel(const float4* __restrict__ src, float4* __restrict__ dst, int n) {
    int i = blockIdx.x * blockDim.x + threadIdx.x;
    int stride = gridDim.x * blockDim.x;
    for (; i < n; i += stride) dst[i] = src[i];
}

// -------------------------------------------------------------- f32 -> bf16
__global__ void convert_bf16_kernel(const float4* __restrict__ src, unsigned short* __restrict__ dst, int n4) {
    int i = blockIdx.x * blockDim.x + threadIdx.x;
    int stride = gridDim.x * blockDim.x;
    for (; i < n4; i += stride) {
        float4 f = src[i];
        dst[4*i+0] = f32_to_bf16(f.x);
        dst[4*i+1] = f32_to_bf16(f.y);
        dst[4*i+2] = f32_to_bf16(f.z);
        dst[4*i+3] = f32_to_bf16(f.w);
    }
}

// ------------------------------------------------- shared 128x128 GEMM tile
// C(128x128) = A(128xK, bf16, row-major) * B(Kx128, f32, row-major).
// A is streamed with async global->LDS DMA, double-buffered; B is converted
// f32->bf16 on the VALU into a transposed (N-major) LDS tile.
__device__ __forceinline__ void gemm128_tile(const unsigned short* __restrict__ A, int lda,
                                             const float* __restrict__ B, int ldb,
                                             float* __restrict__ C, int ldc, int Kdim)
{
    __shared__ unsigned short sA[2][128][40];   // [m][k], padded stride 40
    __shared__ unsigned short sB[2][128][40];   // [n][k] (transposed), stride 40
    const int tid  = threadIdx.x;
    const int lane = tid & 31;
    const int wid  = tid >> 5;
    const int wm   = (wid & 3) * 32;   // 4 waves along M
    const int wn   = (wid >> 2) * 64;  // 2 waves along N
    const int half = lane >> 4, ln = lane & 15;

    // per-thread fill roles (fixed across iterations)
    const int arow = tid >> 1, ahf = (tid & 1) * 16;      // A: 2 threads/row, 32B each
    const int bk   = tid >> 3, bn0 = (tid & 7) * 16;      // B: 8 threads/k-row

    auto fillA = [&](int buf, int kk) {
        const unsigned short* gsrc = A + (size_t)arow * lda + kk + ahf;
        unsigned ldst = (unsigned)(size_t)&sA[buf][arow][ahf];
        async_load_b128(ldst,      gsrc);
        async_load_b128(ldst + 16, gsrc + 8);
    };
    auto fillB = [&](int buf, int kk) {
        const float* src = B + (size_t)(kk + bk) * ldb + bn0;
        #pragma unroll
        for (int j = 0; j < 4; ++j) {
            float4 f = *(const float4*)(src + 4 * j);
            sB[buf][bn0 + 4*j + 0][bk] = f32_to_bf16(f.x);
            sB[buf][bn0 + 4*j + 1][bk] = f32_to_bf16(f.y);
            sB[buf][bn0 + 4*j + 2][bk] = f32_to_bf16(f.z);
            sB[buf][bn0 + 4*j + 3][bk] = f32_to_bf16(f.w);
        }
    };

    v8f acc[2][4];
    #pragma unroll
    for (int i = 0; i < 2; ++i)
        #pragma unroll
        for (int j = 0; j < 4; ++j) acc[i][j] = {};

    const int nk = Kdim / 32;
    fillA(0, 0);
    fillB(0, 0);

    for (int it = 0; it < nk; ++it) {
        const int cur = it & 1;
        const bool hasnext = (it + 1 < nk);
        if (hasnext) {             // prefetch next tile into the other buffer
            fillA(cur ^ 1, (it + 1) * 32);
            fillB(cur ^ 1, (it + 1) * 32);
            wait_async_le2();      // older (current-tile) async pair complete
        } else {
            wait_async_le0();
        }
        __syncthreads();           // current buffers visible to all waves

        v16bf aF[2], bF[4];
        #pragma unroll
        for (int i = 0; i < 2; ++i) aF[i] = load_frag_a(&sA[cur][wm + 16*i][0], 40, lane);
        #pragma unroll
        for (int j = 0; j < 4; ++j) bF[j] = load_frag_b(&sB[cur][wn + 16*j][0], 40, lane);
        #pragma unroll
        for (int i = 0; i < 2; ++i)
            #pragma unroll
            for (int j = 0; j < 4; ++j)
                acc[i][j] = wmma_bf16(aF[i], bF[j], acc[i][j]);

        __syncthreads();           // all waves done reading: buffer reusable
    }

    #pragma unroll
    for (int i = 0; i < 2; ++i)
        #pragma unroll
        for (int j = 0; j < 4; ++j)
            #pragma unroll
            for (int r = 0; r < 8; ++r) {
                int m = wm + 16*i + half*8 + r;
                int n = wn + 16*j + ln;
                C[(size_t)m * ldc + n] = acc[i][j][r];
            }
}

// ---------------------------------------------------------------- QKV GEMM
// qkvraw layout: [hc][t][h], hc = 0..31 q-heads, 32..39 k-heads, 40..47 v-heads
__global__ __launch_bounds__(256) void qkv_gemm_kernel(
    const unsigned short* __restrict__ xb, const float* __restrict__ wq,
    const float* __restrict__ wk,          const float* __restrict__ wv,
    float* __restrict__ qkvraw)
{
    int m0 = blockIdx.x * 128;
    int hc = blockIdx.y;
    const float* B;
    if (hc < 32)      B = wq + (size_t)hc        * DD * HH;
    else if (hc < 40) B = wk + (size_t)(hc - 32) * DD * HH;
    else              B = wv + (size_t)(hc - 40) * DD * HH;
    gemm128_tile(xb + (size_t)m0 * DD, DD,
                 B, HH,
                 qkvraw + ((size_t)hc * TT + m0) * HH, HH, DD);
}

// -------------------------------------------- RoPE + cache scatter + repack
__global__ __launch_bounds__(64) void rope_kernel(
    const float* __restrict__ qkvraw,
    const int* __restrict__ positions, const int* __restrict__ write_idx,
    float* __restrict__ kc_out, float* __restrict__ vc_out,
    unsigned short* __restrict__ qb, unsigned short* __restrict__ kb,
    unsigned short* __restrict__ vb)
{
    int t  = blockIdx.x;
    int hc = blockIdx.y;
    int j  = threadIdx.x;                 // 0..63 (pair j, j+64)
    const float* raw = qkvraw + ((size_t)hc * TT + t) * HH;
    float x1 = raw[j], x2 = raw[j + 64];

    if (hc < 40) {  // rope for q and k
        float pos = (float)positions[t];
        float inv_freq = powf(500000.0f, -((float)(2 * j)) / 128.0f);
        float ang = pos * inv_freq;
        float c = cosf(ang), s = sinf(ang);
        float o1 = x1 * c - x2 * s;
        float o2 = x2 * c + x1 * s;
        if (hc < 32) {
            const float scale = 0.08838834764831845f;  // 1/sqrt(128)
            size_t base = ((size_t)hc * TT + t) * HH;
            qb[base + j]      = f32_to_bf16(o1 * scale);
            qb[base + j + 64] = f32_to_bf16(o2 * scale);
        } else {
            int kh   = hc - 32;
            int slot = write_idx[t];
            size_t cb = ((size_t)slot * KH + kh) * HH;
            kc_out[cb + j]      = o1;
            kc_out[cb + j + 64] = o2;
            size_t base = ((size_t)kh * TT + t) * HH;
            kb[base + j]      = f32_to_bf16(o1);
            kb[base + j + 64] = f32_to_bf16(o2);
        }
    } else {        // v: plain copy
        int kh   = hc - 40;
        int slot = write_idx[t];
        size_t cb = ((size_t)slot * KH + kh) * HH;
        vc_out[cb + j]      = x1;
        vc_out[cb + j + 64] = x2;
        size_t base = ((size_t)kh * TT + t) * HH;
        vb[base + j]      = f32_to_bf16(x1);
        vb[base + j + 64] = f32_to_bf16(x2);
    }
}

// ------------------------------------------------- flash attention (causal)
// block: (T/128 query tiles, 32 heads), 256 threads (8 waves, 16 q-rows each)
__global__ __launch_bounds__(256) void attn_kernel(
    const unsigned short* __restrict__ qb, const unsigned short* __restrict__ kb,
    const unsigned short* __restrict__ vb, unsigned short* __restrict__ attnb)
{
    __shared__ unsigned short sVT[128][72];  // [h][key], stride 72 (16B mult)
    __shared__ unsigned short sP [128][72];  // [qrow][key]

    const int tq0  = blockIdx.x * 128;
    const int head = blockIdx.y;
    const int kh   = head >> 2;              // GQA: 4 q-heads per kv-head
    const int tid  = threadIdx.x;
    const int lane = tid & 31;
    const int wid  = tid >> 5;
    const int m0w  = wid * 16;
    const int half = lane >> 4, ln = lane & 15;

    const unsigned short* qbase  = qb + ((size_t)head * TT + tq0 + m0w) * HH;
    const unsigned short* kbase0 = kb + (size_t)kh * TT * HH;
    const unsigned short* vbase0 = vb + (size_t)kh * TT * HH;

    // Q fragments held in registers for the whole block (A layout, direct from global)
    v16bf qF[4];
    #pragma unroll
    for (int c = 0; c < 4; ++c) qF[c] = load_frag_a(qbase + 32 * c, HH, lane);

    v8f Oa[8];
    float Mrun[8], Lrun[8];
    #pragma unroll
    for (int jh = 0; jh < 8; ++jh) Oa[jh] = {};
    #pragma unroll
    for (int r = 0; r < 8; ++r) { Mrun[r] = NEG_INF; Lrun[r] = 0.0f; }

    const int nkt = (tq0 + 128) / 64;   // causal: key tiles of 64
    for (int kt = 0; kt < nkt; ++kt) {
        const int tk0 = kt * 64;
        __syncthreads();                 // protect sVT from previous iter readers
        {   // load V tile (64 keys x 128 h) transposed into sVT[h][key]
            int key = tid >> 2;          // 0..63
            int h0  = (tid & 3) * 32;
            const unsigned short* src = vbase0 + (size_t)(tk0 + key) * HH + h0;
            #pragma unroll
            for (int jj = 0; jj < 4; ++jj) {
                U4 u; u.q = *(const uint4*)(src + 8 * jj);
                #pragma unroll
                for (int e = 0; e < 8; ++e) sVT[h0 + 8*jj + e][key] = u.s[e];
            }
        }
        __syncthreads();

        // S = Q * K^T  (K B-fragments straight from global: key rows are
        // contiguous in H, exactly the N-major layout the B fragment wants)
        v8f S[4];
        #pragma unroll
        for (int j = 0; j < 4; ++j) S[j] = {};
        const unsigned short* kbase = kbase0 + (size_t)tk0 * HH;
        #pragma unroll
        for (int c = 0; c < 4; ++c) {
            #pragma unroll
            for (int j = 0; j < 4; ++j) {
                v16bf bF = load_frag_b(kbase + (size_t)(16 * j) * HH + 32 * c, HH, lane);
                S[j] = wmma_bf16(qF[c], bF, S[j]);
            }
        }

        // causal mask + online softmax (rows live across half-wave lanes)
        #pragma unroll
        for (int r = 0; r < 8; ++r) {
            int mrow = tq0 + m0w + half * 8 + r;
            #pragma unroll
            for (int j = 0; j < 4; ++j) {
                int ncol = tk0 + 16 * j + ln;
                if (ncol > mrow) S[j][r] = NEG_INF;
            }
            float tm = fmaxf(fmaxf(S[0][r], S[1][r]), fmaxf(S[2][r], S[3][r]));
            #pragma unroll
            for (int off = 8; off >= 1; off >>= 1) tm = fmaxf(tm, __shfl_xor(tm, off));
            float newM = fmaxf(Mrun[r], tm);
            float fsc  = expf(Mrun[r] - newM);
            float rs = 0.0f;
            #pragma unroll
            for (int j = 0; j < 4; ++j) {
                float p = expf(S[j][r] - newM);
                S[j][r] = p;
                rs += p;
            }
            #pragma unroll
            for (int off = 8; off >= 1; off >>= 1) rs += __shfl_xor(rs, off);
            Lrun[r] = Lrun[r] * fsc + rs;
            Mrun[r] = newM;
            #pragma unroll
            for (int jh = 0; jh < 8; ++jh) Oa[jh][r] *= fsc;
            int prow = m0w + half * 8 + r;
            #pragma unroll
            for (int j = 0; j < 4; ++j)
                sP[prow][16 * j + ln] = f32_to_bf16(S[j][r]);
        }
        // sP rows are wave-private: no cross-wave barrier needed here.

        // O += P * V   (A frags from sP, B frags from transposed sVT)
        #pragma unroll
        for (int c = 0; c < 2; ++c) {
            v16bf pF = load_frag_a(&sP[m0w][0] + 32 * c, 72, lane);
            #pragma unroll
            for (int jh = 0; jh < 8; ++jh) {
                v16bf vF = load_frag_b(&sVT[16 * jh][0] + 32 * c, 72, lane);
                Oa[jh] = wmma_bf16(pF, vF, Oa[jh]);
            }
        }
    }

    // epilogue: normalize and write attn[t][head*128 + h] as bf16 (feeds the
    // output projection's async-DMA A path directly)
    #pragma unroll
    for (int r = 0; r < 8; ++r) {
        float inv = 1.0f / Lrun[r];
        int mrow = tq0 + m0w + half * 8 + r;
        #pragma unroll
        for (int jh = 0; jh < 8; ++jh)
            attnb[(size_t)mrow * (NH * HH) + head * HH + 16 * jh + ln] =
                f32_to_bf16(Oa[jh][r] * inv);
    }
}

// ------------------------------------------------------------- output GEMM
__global__ __launch_bounds__(256) void out_gemm_kernel(
    const unsigned short* __restrict__ attnb, const float* __restrict__ wo,
    float* __restrict__ o)
{
    int m0 = blockIdx.x * 128;
    int n0 = blockIdx.y * 128;
    gemm128_tile(attnb + (size_t)m0 * DD, DD,
                 wo + n0, DD,
                 o + (size_t)m0 * DD + n0, DD, DD);
}

// ------------------------------------------------------------------ launch
extern "C" void kernel_launch(void* const* d_in, const int* in_sizes, int n_in,
                              void* d_out, int out_size, void* d_ws, size_t ws_size,
                              hipStream_t stream)
{
    const float* x     = (const float*)d_in[0];
    const float* wq    = (const float*)d_in[1];
    const float* wk    = (const float*)d_in[2];
    const float* wv    = (const float*)d_in[3];
    const float* wo    = (const float*)d_in[4];
    const float* kc_in = (const float*)d_in[5];
    const float* vc_in = (const float*)d_in[6];
    const int* positions = (const int*)d_in[7];
    const int* windices  = (const int*)d_in[8];

    float* kc_out = (float*)d_out;                         // 32768*8*128
    float* vc_out = kc_out + (size_t)SLOTS * KH * HH;
    float* o_out  = vc_out + (size_t)SLOTS * KH * HH;

    uint8_t* ws = (uint8_t*)d_ws;
    float*          qkvraw = (float*)ws;                               // 48*T*H f32  (50.3 MB)
    unsigned short* qb = (unsigned short*)(ws + 50331648);             // 32*T*H bf16 (16.8 MB)
    unsigned short* kb = (unsigned short*)(ws + 50331648 + 16777216);  //  8*T*H bf16 ( 4.2 MB)
    unsigned short* vb = (unsigned short*)(ws + 50331648 + 16777216 + 4194304);
    unsigned short* xb = (unsigned short*)(ws + 75497472);             // T*D bf16    (16.8 MB)
    unsigned short* attnb = xb;  // reuses xb region once qkv_gemm consumed it

    // 1) initialize output caches from input caches (d_out is poisoned)
    const int n4 = (SLOTS * KH * HH) / 4;
    copy_f4_kernel<<<4096, 256, 0, stream>>>((const float4*)kc_in, (float4*)kc_out, n4);
    copy_f4_kernel<<<4096, 256, 0, stream>>>((const float4*)vc_in, (float4*)vc_out, n4);

    // 2) x -> bf16 (activations feed the async-DMA A path of the GEMMs)
    convert_bf16_kernel<<<2048, 256, 0, stream>>>((const float4*)x, xb, (TT * DD) / 4);

    // 3) QKV projections (bf16 WMMA, f32 accumulate, async double-buffered)
    qkv_gemm_kernel<<<dim3(TT / 128, 48), 256, 0, stream>>>(xb, wq, wk, wv, qkvraw);

    // 4) RoPE + cache scatter + bf16 repack
    rope_kernel<<<dim3(TT, 48), 64, 0, stream>>>(qkvraw, positions, windices,
                                                 kc_out, vc_out, qb, kb, vb);

    // 5) causal GQA flash attention
    attn_kernel<<<dim3(TT / 128, NH), 256, 0, stream>>>(qb, kb, vb, attnb);

    // 6) output projection
    out_gemm_kernel<<<dim3(TT / 128, DD / 128), 256, 0, stream>>>(attnb, wo, o_out);
}
